// model_84293028151836
// MI455X (gfx1250) — matrix operand order repaired and checked by reading, not compiled, not run on hardware
//
#include <hip/hip_runtime.h>
#include <hip/hip_bf16.h>
#include <math.h>

typedef __attribute__((ext_vector_type(16))) _Float16 v16h;
typedef __attribute__((ext_vector_type(8)))  _Float16 v8h;
typedef __attribute__((ext_vector_type(8)))  float    v8f;
typedef __attribute__((ext_vector_type(4)))  int      v4i;

#define N_MEN   1200
#define P_PAIRS 719400            // N*(N-1)/2
#define HID     128
#define RANK    64
#define KX      39
#define WSTRIDE 1216              // padded cols of W (zeros beyond 1199)
#define VROWS   1216              // padded rows of V (zeros beyond 1199)
#define SDP_ITERS 100
#define MOMB    512               // blocks for moment pass (fixed, used in reducer)

// ---- workspace layout (bytes, all 16B aligned) ----
#define O_CP    0                                  // [MOMB][820] f32 partials (780 C pairs + 39 S1)
#define O_BNA   1679360                            // 128 f32
#define O_BNC   1679872                            // 128 f32
#define O_WSQP  1680384                            // per-block |w|^2 partials
#define O_LR    1725696                            // 1 f32
#define O_W1H   1725952                            // W1^T f16 [128][64]
#define O_WMAT  1742336                            // W f16 [1200][1216]
#define O_VA    (1742336 + 2918400)                // V row-major f16 [1216][64]
#define O_VB    (O_VA + 155648)
#define O_VAT   (O_VB + 155648)                    // V^T f16 [64][1216]
#define O_VBT   (O_VAT + 155648)

// -------- async global->LDS (CDNA5) with fallback ---------------------------
#if defined(__gfx1250__) && __has_builtin(__builtin_amdgcn_global_load_async_to_lds_b128) && __has_builtin(__builtin_amdgcn_s_wait_asynccnt)
#define USE_ASYNC_LDS 1
#endif

__device__ inline void copy16_g2l(void* lds_dst, const void* gsrc) {
#ifdef USE_ASYNC_LDS
  // clang-22 signature (from diagnostics): (v4i AS1* src, v4i AS3* dst, imm offset, imm cpol)
  __builtin_amdgcn_global_load_async_to_lds_b128(
      (__attribute__((address_space(1))) v4i*)gsrc,
      (__attribute__((address_space(3))) v4i*)lds_dst, 0, 0);
#else
  *(v8h*)lds_dst = *(const v8h*)gsrc;
#endif
}

__device__ inline void copy16_wait_all() {
#ifdef USE_ASYNC_LDS
  __builtin_amdgcn_s_wait_asynccnt(0);
#endif
}

// ================= WMMA fragment helpers (CDNA5 documented layouts) =========
// A (16x32 f16, MxK): lane&15 = M row; lanes>=16 shift K by 8.
// Per lane the fragment is { base[0..7], base[16..23] } with base = row + kb.
__device__ inline v16h make_a(const _Float16* base) {        // base 16B aligned
  v8h lo = *(const v8h*)base;
  v8h hi = *(const v8h*)(base + 16);
  return __builtin_shufflevector(lo, hi, 0,1,2,3,4,5,6,7,8,9,10,11,12,13,14,15);
}
// B (32x16 f16, KxN) read from a K-contiguous (transposed) buffer:
// per lane the fragment is { base[0..7], base[16..23] } with base = column + kb (same K order as A).
__device__ inline v16h make_b(const _Float16* base) {        // base 16B aligned
  v8h lo = *(const v8h*)base;
  v8h hi = *(const v8h*)(base + 16);
  return __builtin_shufflevector(lo, hi, 0,1,2,3,4,5,6,7,8,9,10,11,12,13,14,15);
}

// A fragment from f32 x with K bound (=39) and row bound (edge pass only)
__device__ inline v16h load_a_x39(const float* __restrict__ x, int r0, int lane, int koff) {
  int m = lane & 15;
  int kb = (lane >> 4) * 8;
  int p = r0 + m;
  v16h a;
#pragma unroll
  for (int i = 0; i < 8; ++i) {
    int hh = i >> 2, v = i & 3;
    int k0 = koff + kb + 16 * hh + 2 * v;
    float e0 = 0.f, e1 = 0.f;
    if (p < P_PAIRS) {
      if (k0 < KX)     e0 = x[p * KX + k0];
      if (k0 + 1 < KX) e1 = x[p * KX + k0 + 1];
    }
    a[2 * i]     = (_Float16)e0;
    a[2 * i + 1] = (_Float16)e1;
  }
  return a;
}

// ============================ kernels =======================================
__global__ void k_init(char* ws) {
  unsigned idx = blockIdx.x * blockDim.x + threadIdx.x;
  unsigned stride = gridDim.x * blockDim.x;
  unsigned* uw = (unsigned*)(ws + O_WMAT);          // W + VA + VB + VAT + VBT contiguous
  unsigned tot = (2918400u + 155648u * 4u) / 4u;
  for (unsigned i = idx; i < tot; i += stride) uw[i] = 0u;
}

// pass 1: per-block partial X^T X (upper-tri pairs) and column sums of x
__global__ void __launch_bounds__(256) k_moments(const float* __restrict__ x, char* ws) {
  __shared__ float xs[64 * 40];
  int tid = threadIdx.x;
  int qi[4], qj[4]; int nq = 0;
  for (int q = tid; q < 780; q += 256) {
    int i = 0, off = 0;
    while (off + (KX - i) <= q) { off += KX - i; ++i; }
    qi[nq] = i; qj[nq] = i + (q - off); ++nq;
  }
  float acc[4] = {0.f, 0.f, 0.f, 0.f};
  float accs = 0.f;
  long chunk = ((long)P_PAIRS + gridDim.x - 1) / gridDim.x;
  long rbeg = (long)blockIdx.x * chunk;
  long rend = rbeg + chunk; if (rend > (long)P_PAIRS) rend = P_PAIRS;
  for (long t0 = rbeg; t0 < rend; t0 += 64) {
    for (int idx = tid; idx < 64 * KX; idx += 256) {
      int r = idx / KX, c = idx % KX;
      long row = t0 + r;
      xs[r * 40 + c] = (row < rend) ? x[row * KX + c] : 0.f;
    }
    __syncthreads();
    for (int r = 0; r < 64; ++r) {
      const float* xr = &xs[r * 40];
#pragma unroll
      for (int u = 0; u < 4; ++u)
        if (u < nq) acc[u] += xr[qi[u]] * xr[qj[u]];
      if (tid < KX) accs += xr[tid];
    }
    __syncthreads();
  }
  float* cp = (float*)(ws + O_CP) + (long)blockIdx.x * 820;
  for (int u = 0; u < nq; ++u) cp[tid + 256 * u] = acc[u];
  if (tid < KX) cp[780 + tid] = accs;
}

// reduce partials (fixed order), fold BN into per-channel affine, build f16 W1^T (K padded to 64)
__global__ void __launch_bounds__(128) k_bnprep(const float* __restrict__ W1, const float* __restrict__ b1,
                                                const float* __restrict__ gamma, const float* __restrict__ beta,
                                                char* ws) {
  __shared__ float Cs[780];
  __shared__ float ms[KX];
  int tid = threadIdx.x;
  const float* cp = (const float*)(ws + O_CP);
  const float invP = 1.0f / (float)P_PAIRS;
  for (int e = tid; e < 819; e += 128) {
    float s = 0.f;
    for (int b = 0; b < MOMB; ++b) s += cp[(long)b * 820 + e];
    if (e < 780) Cs[e] = s; else ms[e - 780] = s * invP;
  }
  __syncthreads();
  float* bnA = (float*)(ws + O_BNA);
  float* bnC = (float*)(ws + O_BNC);
  _Float16* W1hT = (_Float16*)(ws + O_W1H);
  int j = tid;  // 128 channels, 128 threads
  float wj[KX];
  float mu_u = 0.f;
  for (int i = 0; i < KX; ++i) { wj[i] = W1[i * HID + j]; mu_u += ms[i] * wj[i]; }
  float q = 0.f;
  for (int i = 0; i < KX; ++i) {
    float ti = 0.f;
    for (int k = 0; k < KX; ++k) {
      int lo = i < k ? i : k, hi = i < k ? k : i;
      ti += Cs[lo * KX - lo * (lo - 1) / 2 + (hi - lo)] * wj[k];
    }
    q += ti * wj[i];
  }
  q *= invP;
  float var = q - mu_u * mu_u;
  float aj = gamma[j] * rsqrtf(var + 1e-5f);
  bnA[j] = aj;
  bnC[j] = beta[j] - aj * mu_u;   // b1 cancels into this affine term
  for (int k = 0; k < 64; ++k)
    W1hT[j * 64 + k] = (_Float16)((k < KX) ? W1[k * HID + j] : 0.f);
  (void)b1;
}

// pass 2: fused x@W1 (WMMA) -> BN affine -> ReLU -> dot W2 -> scatter symmetric f16 W; |w|^2 partials
__global__ void __launch_bounds__(128) k_edge(const float* __restrict__ x, const float* __restrict__ W2,
                                              const float* __restrict__ b2, char* ws) {
  __shared__ float hbuf[4][16 * HID];
  __shared__ float red[128];
  const float* bnA = (const float*)(ws + O_BNA);
  const float* bnC = (const float*)(ws + O_BNC);
  const _Float16* W1hT = (const _Float16*)(ws + O_W1H);
  _Float16* Wm = (_Float16*)(ws + O_WMAT);
  int tid = threadIdx.x, lane = tid & 31, wv = tid >> 5;
  int strip = blockIdx.x * 4 + wv;
  int r0 = strip * 16;
  v16h a0 = load_a_x39(x, r0, lane, 0);
  v16h a1 = load_a_x39(x, r0, lane, 32);
  int n_ = lane & 15;
  int mb = (lane >> 4) * 8;
  int kb16 = (lane >> 4) * 8;
#pragma unroll
  for (int nt = 0; nt < 8; ++nt) {
    const _Float16* bb = W1hT + (nt * 16 + n_) * 64 + kb16;
    v16h b0 = make_b(bb);
    v16h b1f = make_b(bb + 32);
    v8f c = {};
    c = __builtin_amdgcn_wmma_f32_16x16x32_f16(false, a0, false, b0, (short)0, c, false, false);
    c = __builtin_amdgcn_wmma_f32_16x16x32_f16(false, a1, false, b1f, (short)0, c, false, false);
    int ch = nt * 16 + n_;
    float ga = bnA[ch], gc = bnC[ch];
#pragma unroll
    for (int g = 0; g < 8; ++g) {
      float hv = ga * c[g] + gc;
      hbuf[wv][(mb + g) * HID + ch] = hv > 0.f ? hv : 0.f;
    }
  }
  __syncthreads();
  float wsql = 0.f;
  if (lane < 16) {
    long p = (long)r0 + lane;
    float s = b2[0];
    for (int ch = 0; ch < HID; ++ch) s += hbuf[wv][lane * HID + ch] * W2[ch];
    if (p < (long)P_PAIRS) {
      // invert flat upper-triangle index p -> (i, j), j > i
      double dp = (double)p;
      double t = 2.0 * N_MEN - 1.0;
      int i = (int)((t - sqrt(t * t - 8.0 * dp)) * 0.5);
      if (i < 0) i = 0; if (i > N_MEN - 2) i = N_MEN - 2;
      while ((long)i * (2 * N_MEN - i - 1) / 2 > p) --i;
      while ((long)(i + 1) * (2 * N_MEN - i - 2) / 2 <= p) ++i;
      long j = p - (long)i * (2 * N_MEN - i - 1) / 2 + i + 1;
      _Float16 wh = (_Float16)s;
      Wm[(long)i * WSTRIDE + j] = wh;
      Wm[(long)j * WSTRIDE + i] = wh;
      wsql = s * s;
    }
  }
  red[tid] = wsql;
  __syncthreads();
  if (tid == 0) {
    float tot = 0.f;
    for (int i = 0; i < 128; ++i) tot += red[i];
    ((float*)(ws + O_WSQP))[blockIdx.x] = tot;
  }
}

__global__ void __launch_bounds__(256) k_lr(char* ws, int nblk) {
  __shared__ float red[256];
  const float* wp = (const float*)(ws + O_WSQP);
  float s = 0.f;
  for (int i = threadIdx.x; i < nblk; i += 256) s += wp[i];
  red[threadIdx.x] = s;
  __syncthreads();
  if (threadIdx.x == 0) {
    float t = 0.f;
    for (int i = 0; i < 256; ++i) t += red[i];
    *(float*)(ws + O_LR) = 1.0f / (sqrtf(2.0f * t) + 1e-8f);   // ||W||_F = sqrt(2*sum w^2)
  }
}

__global__ void k_vinit(const float* __restrict__ V0, char* ws) {
  _Float16* Va = (_Float16*)(ws + O_VA);
  _Float16* VaT = (_Float16*)(ws + O_VAT);
  int r = blockIdx.x * blockDim.x + threadIdx.x;
  if (r < N_MEN) {
    float s = 0.f;
    for (int c = 0; c < RANK; ++c) { float v = V0[r * RANK + c]; s += v * v; }
    float rn = rsqrtf(s);
    for (int c = 0; c < RANK; ++c) {
      _Float16 hv = (_Float16)(V0[r * RANK + c] * rn);
      Va[r * RANK + c] = hv;
      VaT[c * VROWS + r] = hv;
    }
  }
}

// one Burer–Monteiro step: V' = rownorm(V + lr * W @ V)
// A strip of W staged to LDS via async global->LDS; B operands from V^T (contiguous)
__global__ void __launch_bounds__(128) k_sdp_step(const _Float16* __restrict__ VinT,
                                                  _Float16* __restrict__ VoutR,
                                                  _Float16* __restrict__ VoutT, char* ws) {
  __shared__ _Float16 wlds[16 * WSTRIDE];   // 38912 B, 16-row strip of W
  __shared__ float vn[16 * RANK];
  __shared__ float rn[16];
  const _Float16* Wm = (const _Float16*)(ws + O_WMAT);
  const float lr = *(const float*)(ws + O_LR);
  int tid = threadIdx.x, lane = tid & 31, wv = tid >> 5;
  int r0 = blockIdx.x * 16;

  // stage 16x1216 f16 strip (rows are contiguous): 2432 x 16B, 19 per thread
  {
    const _Float16* gsrc = Wm + (long)r0 * WSTRIDE;
#pragma unroll 1
    for (int it = 0; it < 19; ++it) {
      int idx = tid + it * 128;
      copy16_g2l(&wlds[idx * 8], gsrc + idx * 8);
    }
    copy16_wait_all();
  }
  __syncthreads();

  int n_ = lane & 15, mb = (lane >> 4) * 8;
  int kb = (lane >> 4) * 8;      // A fragment K base
  int kb16 = (lane >> 4) * 8;    // B fragment K base
  int m = lane & 15;
  const _Float16* bcol = VinT + (long)(wv * 16 + n_) * VROWS;

  v8f acc = {};
  for (int kk = 0; kk < WSTRIDE; kk += 32) {
    if (kk + 64 < WSTRIDE) __builtin_prefetch(bcol + kk + 64, 0, 1);
    v16h a = make_a(&wlds[m * WSTRIDE + kk + kb]);
    v16h b = make_b(bcol + kk + kb16);
    acc = __builtin_amdgcn_wmma_f32_16x16x32_f16(false, a, false, b, (short)0, acc, false, false);
  }

  // V + lr*(W@V): read old V values contiguously from V^T (8 halfs per lane)
  v8h vold = *(const v8h*)(bcol + r0 + mb);
#pragma unroll
  for (int g = 0; g < 8; ++g)
    vn[(mb + g) * RANK + wv * 16 + n_] = (float)vold[g] + lr * acc[g];
  __syncthreads();
  if (tid < 16) {
    float s = 0.f;
    for (int c = 0; c < RANK; ++c) { float v = vn[tid * RANK + c]; s += v * v; }
    rn[tid] = rsqrtf(s);
  }
  __syncthreads();
  for (int idx = tid; idx < 16 * RANK; idx += 128) {
    int mm = idx >> 6, c = idx & 63;
    _Float16 hv = (_Float16)(vn[idx] * rn[mm]);
    VoutR[(long)(r0 + mm) * RANK + c] = hv;
    VoutT[(long)c * VROWS + (r0 + mm)] = hv;
  }
}

// X = clip(V V^T, 0, 1)   (A and B both from row-major V; both contiguous)
__global__ void __launch_bounds__(128) k_xout(char* ws, float* __restrict__ out) {
  const _Float16* V = (const _Float16*)(ws + O_VA);
  int tid = threadIdx.x, lane = tid & 31, wv = tid >> 5;
  int nt = blockIdx.y * 4 + wv;
  if (nt >= 75) return;                 // wave-uniform -> EXEC all-ones for active waves
  int r0 = blockIdx.x * 16, c0 = nt * 16;
  int m = lane & 15, kb = (lane >> 4) * 8;
  int n_ = lane & 15, kb16 = (lane >> 4) * 8, mb = (lane >> 4) * 8;
  const _Float16* arow = V + (long)(r0 + m) * RANK;
  const _Float16* bcol = V + (long)(c0 + n_) * RANK;
  v16h a0 = make_a(arow + kb);
  v16h a1 = make_a(arow + 32 + kb);
  v16h b0 = make_b(bcol + kb16);
  v16h b1 = make_b(bcol + 32 + kb16);
  v8f c = {};
  c = __builtin_amdgcn_wmma_f32_16x16x32_f16(false, a0, false, b0, (short)0, c, false, false);
  c = __builtin_amdgcn_wmma_f32_16x16x32_f16(false, a1, false, b1, (short)0, c, false, false);
#pragma unroll
  for (int g = 0; g < 8; ++g) {
    float v = c[g];
    v = v < 0.f ? 0.f : (v > 1.f ? 1.f : v);
    out[(long)(r0 + mb + g) * N_MEN + c0 + n_] = v;
  }
}

extern "C" void kernel_launch(void* const* d_in, const int* in_sizes, int n_in,
                              void* d_out, int out_size, void* d_ws, size_t ws_size,
                              hipStream_t stream) {
  (void)in_sizes; (void)n_in; (void)out_size; (void)ws_size;
  const float* x     = (const float*)d_in[0];
  const float* W1    = (const float*)d_in[1];
  const float* b1    = (const float*)d_in[2];
  const float* gamma = (const float*)d_in[3];
  const float* beta  = (const float*)d_in[4];
  const float* W2    = (const float*)d_in[5];
  const float* b2    = (const float*)d_in[6];
  const float* V0    = (const float*)d_in[7];
  char* ws = (char*)d_ws;
  float* out = (float*)d_out;

  k_init<<<512, 256, 0, stream>>>(ws);
  k_moments<<<MOMB, 256, 0, stream>>>(x, ws);
  k_bnprep<<<1, 128, 0, stream>>>(W1, b1, gamma, beta, ws);
  int strips = (P_PAIRS + 15) / 16;        // 44963
  int eb = (strips + 3) / 4;               // 11241 blocks, 4 waves each
  k_edge<<<eb, 128, 0, stream>>>(x, W2, b2, ws);
  k_lr<<<1, 256, 0, stream>>>(ws, eb);
  k_vinit<<<(N_MEN + 63) / 64, 64, 0, stream>>>(V0, ws);

  _Float16* VaR = (_Float16*)(ws + O_VA);
  _Float16* VbR = (_Float16*)(ws + O_VB);
  _Float16* VaT = (_Float16*)(ws + O_VAT);
  _Float16* VbT = (_Float16*)(ws + O_VBT);
  for (int t = 0; t < SDP_ITERS; ++t) {
    const _Float16* inT = (t & 1) ? (const _Float16*)VbT : (const _Float16*)VaT;
    _Float16* oR = (t & 1) ? VaR : VbR;
    _Float16* oT = (t & 1) ? VaT : VbT;
    k_sdp_step<<<75, 128, 0, stream>>>(inT, oR, oT, ws);
  }
  // after an even number of iterations the current V lives in VA / VAT
  k_xout<<<dim3(75, 19), 128, 0, stream>>>(ws, out);
}
